// OTAlignmentLoss_26972394619275
// MI455X (gfx1250) — matrix operand order repaired and checked by reading, not compiled, not run on hardware
//
#include <hip/hip_runtime.h>
#include <hip/hip_bf16.h>

// ---------------------------------------------------------------------------
// OT alignment loss on MI455X (gfx1250, wave32).
//   C (64MB) and K=exp(-20C) (64MB) both live in the 192MB L2 -> Sinkhorn
//   matvecs are L2-resident.  Both big GEMMs use v_wmma_f32_16x16x32_bf16
//   with 16x64 register blocking per wave (A fragments reused across 4 B
//   tiles).  The cost GEMM uses a hi/lo bf16 split (3 WMMAs) for ~fp32
//   accuracy since exp(-20*C) amplifies error in C.
// ---------------------------------------------------------------------------

typedef __attribute__((ext_vector_type(16))) __bf16 bf16x16;
typedef __attribute__((ext_vector_type(8)))  __bf16 bf16x8;
typedef __attribute__((ext_vector_type(8)))  float  f32x8;
typedef __attribute__((ext_vector_type(4)))  float  f32x4;

#define N_PTS 4096
#define DIMS  256
#define INV_EPS 20.0f        // 1/0.05
#define MU (1.0f / 4096.0f)

__device__ __forceinline__ __bf16 f2bf(float f) {
    unsigned u = __builtin_bit_cast(unsigned, f);
    unsigned r = u + 0x7FFFu + ((u >> 16) & 1u);   // round-to-nearest-even
    unsigned short h = (unsigned short)(r >> 16);
    return __builtin_bit_cast(__bf16, h);
}
__device__ __forceinline__ float bf2f(__bf16 b) {
    unsigned u = ((unsigned)__builtin_bit_cast(unsigned short, b)) << 16;
    return __builtin_bit_cast(float, u);
}
__device__ __forceinline__ float wave_sum(float v) {
#pragma unroll
    for (int off = 16; off > 0; off >>= 1) v += __shfl_xor(v, off, 32);
    return v;
}
__device__ __forceinline__ float wave_min(float v) {
#pragma unroll
    for (int off = 16; off > 0; off >>= 1) v = fminf(v, __shfl_xor(v, off, 32));
    return v;
}

// A fragment (16x32 bf16, row-major source, row stride ld):
//   lane<16: K = {0..7, 16..23}; lane>=16: K = {8..15, 24..31}
__device__ __forceinline__ bf16x16 load_fragA(const __bf16* base, int row0, int ld, int kbase) {
    int lane  = threadIdx.x & 31;
    int khalf = lane >> 4;
    const __bf16* p = base + (size_t)(row0 + (lane & 15)) * ld + kbase + khalf * 8;
    bf16x8 lo = *(const bf16x8*)p;
    bf16x8 hi = *(const bf16x8*)(p + 16);
    bf16x16 a;
#pragma unroll
    for (int t = 0; t < 8; ++t) { a[t] = lo[t]; a[8 + t] = hi[t]; }
    return a;
}

// B fragment (32x16 bf16).  B[k,n] comes from row (col0+n) of a row-major
// matrix (rows = the K direction of the product).  lane<16: K={0..7,16..23},
// lane>=16: K={8..15,24..31}  -> two 16B runs per lane.
__device__ __forceinline__ bf16x16 load_fragB(const __bf16* base, int col0, int ld, int kbase) {
    int lane = threadIdx.x & 31;
    const __bf16* p = base + (size_t)(col0 + (lane & 15)) * ld + kbase + (lane >> 4) * 8;
    bf16x8 lo = *(const bf16x8*)p;
    bf16x8 hi = *(const bf16x8*)(p + 16);
    bf16x16 b;
#pragma unroll
    for (int t = 0; t < 8; ++t) { b[t] = lo[t]; b[8 + t] = hi[t]; }
    return b;
}

__device__ __forceinline__ f32x8 wmma_bf16(bf16x16 a, bf16x16 b, f32x8 c) {
    return __builtin_amdgcn_wmma_f32_16x16x32_bf16(false, a, false, b, (short)0, c, false, false);
}

// --- setup: bf16 hi/lo split of src/tgt, bf16 transpose of tgt, u=v=1 -------
__global__ void k_pack(const float* __restrict__ src, const float* __restrict__ tgt,
                       __bf16* src_hi, __bf16* src_lo,
                       __bf16* tgt_hi, __bf16* tgt_lo,
                       __bf16* tgtT, float* u, float* v) {
    int e = blockIdx.x * blockDim.x + threadIdx.x;     // < N_PTS*DIMS
    float s  = src[e];
    __bf16 sh = f2bf(s);
    src_hi[e] = sh;
    src_lo[e] = f2bf(s - bf2f(sh));
    float t  = tgt[e];
    __bf16 th = f2bf(t);
    tgt_hi[e] = th;
    tgt_lo[e] = f2bf(t - bf2f(th));
    int j = e >> 8;            // row of tgt
    int d = e & 255;           // feature index
    tgtT[(size_t)d * N_PTS + j] = th;
    if (e < N_PTS) { u[e] = 1.0f; v[e] = 1.0f; }
}

// --- row squared norms of src and tgt (one wave per row) --------------------
__global__ void k_norms(const float* __restrict__ src, const float* __restrict__ tgt,
                        float* x2, float* y2) {
    int wave = (blockIdx.x * blockDim.x + threadIdx.x) >> 5;   // 0..8191
    int lane = threadIdx.x & 31;
    const float* base = (wave < N_PTS) ? src + (size_t)wave * DIMS
                                       : tgt + (size_t)(wave - N_PTS) * DIMS;
    float acc = 0.f;
    for (int k = lane; k < DIMS; k += 32) { float a = base[k]; acc += a * a; }
    acc = wave_sum(acc);
    if (lane == 0) { if (wave < N_PTS) x2[wave] = acc; else y2[wave - N_PTS] = acc; }
}

// --- C = max(0, x2 + y2 - 2*src.tgt^T)  and  K = exp(-C/eps) ----------------
// One wave per 16x64 strip (4 tiles): A hi/lo fragments reused across 4 B
// tiles -> 12 WMMAs per 10 b128-load-pairs per k-step.
__global__ void k_gemm_C(const __bf16* __restrict__ src_hi, const __bf16* __restrict__ src_lo,
                         const __bf16* __restrict__ tgt_hi, const __bf16* __restrict__ tgt_lo,
                         const float* __restrict__ x2, const float* __restrict__ y2,
                         float* __restrict__ C, float* __restrict__ Km) {
    int wave = blockIdx.x * 8 + (threadIdx.x >> 5);    // 0..16383
    int lane = threadIdx.x & 31;
    int i0 = (wave >> 6) * 16;          // 256 row tiles
    int j0 = (wave & 63) * 64;          // 64 col strips of 4 tiles
    f32x8 acc[4] = {};
#pragma unroll 2
    for (int kb = 0; kb < DIMS; kb += 32) {
        bf16x16 ah = load_fragA(src_hi, i0, DIMS, kb);
        bf16x16 al = load_fragA(src_lo, i0, DIMS, kb);
#pragma unroll
        for (int c = 0; c < 4; ++c) {
            bf16x16 bh = load_fragB(tgt_hi, j0 + c * 16, DIMS, kb);
            bf16x16 bl = load_fragB(tgt_lo, j0 + c * 16, DIMS, kb);
            acc[c] = wmma_bf16(ah, bh, acc[c]);
            acc[c] = wmma_bf16(ah, bl, acc[c]);
            acc[c] = wmma_bf16(al, bh, acc[c]);
        }
    }
    int rbase = i0 + ((lane >> 4) << 3);
#pragma unroll
    for (int c = 0; c < 4; ++c) {
        int col = j0 + c * 16 + (lane & 15);
        float y2c = y2[col];
#pragma unroll
        for (int r = 0; r < 8; ++r) {
            int row = rbase + r;
            float cv = fmaxf(x2[row] + y2c - 2.0f * acc[c][r], 0.0f);
            size_t idx = (size_t)row * N_PTS + col;
            C[idx]  = cv;
            Km[idx] = __expf(-cv * INV_EPS);
        }
    }
}

// --- row minima of C (one wave per row) -------------------------------------
__global__ void k_rowmin(const float* __restrict__ C, float* row_min) {
    int wave = blockIdx.x * 8 + (threadIdx.x >> 5);
    int lane = threadIdx.x & 31;
    const float* rp = C + (size_t)wave * N_PTS;
    float m = 3.4e38f;
    for (int j = lane; j < N_PTS; j += 32) m = fminf(m, rp[j]);
    m = wave_min(m);
    if (lane == 0) row_min[wave] = m;
}

// --- Sinkhorn: u[i] = mu / (sum_j K[i,j] v[j] + 1e-8)  (one wave per row) ---
__global__ void k_u_update(const float* __restrict__ Km, const float* __restrict__ v,
                           float* __restrict__ u) {
    int wave = blockIdx.x * 8 + (threadIdx.x >> 5);
    int lane = threadIdx.x & 31;
    const float* rp = Km + (size_t)wave * N_PTS;
    float acc = 0.f;
    for (int j = lane; j < N_PTS; j += 32) {
        __builtin_prefetch(rp + j + 512, 0, 0);   // global_prefetch_b8
        acc += rp[j] * v[j];
    }
    acc = wave_sum(acc);
    if (lane == 0) u[wave] = MU / (acc + 1e-8f);
}

// --- column-sum partials:  part[by][j] = sum over 128 rows of K[i,j]*u[i] ---
__global__ void k_colsum(const float* __restrict__ Km, const float* __restrict__ u,
                         float* __restrict__ part) {
    int j  = blockIdx.x * 256 + threadIdx.x;
    int i0 = blockIdx.y * 128;
    float acc = 0.f;
#pragma unroll 4
    for (int ii = 0; ii < 128; ++ii) {
        int i = i0 + ii;
        __builtin_prefetch(&Km[(size_t)(i + 8) * N_PTS + j], 0, 0);
        acc += Km[(size_t)i * N_PTS + j] * u[i];
    }
    part[(size_t)blockIdx.y * N_PTS + j] = acc;
}

__global__ void k_v_update(const float* __restrict__ part, float* __restrict__ v) {
    int j = blockIdx.x * 256 + threadIdx.x;
    float s = 0.f;
#pragma unroll
    for (int c = 0; c < 32; ++c) s += part[(size_t)c * N_PTS + j];
    v[j] = MU / (s + 1e-8f);
}

// --- matched = n_s*(u.K.v)@tgt fused with MSE vs src ------------------------
// One wave per 16x64 strip of `matched`: the (expensive) scaled-A fragment is
// built once per k-step and reused across 4 B tiles / 4 WMMAs.
__global__ void k_matched_mse(const float* __restrict__ Km, const float* __restrict__ u,
                              const float* __restrict__ v, const __bf16* __restrict__ tgtT,
                              const float* __restrict__ src, float* __restrict__ mse_part) {
    int wib  = threadIdx.x >> 5;
    int wave = blockIdx.x * 8 + wib;            // 0..1023
    int lane = threadIdx.x & 31;
    int i0 = (wave >> 2) * 16;                  // 256 row tiles
    int d0 = (wave & 3) * 64;                   // 4 col strips of 4 tiles
    int arow  = i0 + (lane & 15);
    int khalf = lane >> 4;
    float rscale = 4096.0f * u[arow];
    const float* krow = Km + (size_t)arow * N_PTS;
    f32x8 acc[4] = {};
    for (int jb = 0; jb < N_PTS; jb += 32) {
        const float* kp = krow + jb + khalf * 8;
        __builtin_prefetch(kp + 512, 0, 0);
        f32x4 k0 = *(const f32x4*)kp;
        f32x4 k1 = *(const f32x4*)(kp + 4);
        f32x4 k2 = *(const f32x4*)(kp + 16);
        f32x4 k3 = *(const f32x4*)(kp + 20);
        const float* vp = v + jb + khalf * 8;
        f32x4 v0 = *(const f32x4*)vp;
        f32x4 v1 = *(const f32x4*)(vp + 4);
        f32x4 v2 = *(const f32x4*)(vp + 16);
        f32x4 v3 = *(const f32x4*)(vp + 20);
        bf16x16 a;
#pragma unroll
        for (int t = 0; t < 4; ++t) {
            a[t]      = f2bf(rscale * k0[t] * v0[t]);
            a[4 + t]  = f2bf(rscale * k1[t] * v1[t]);
            a[8 + t]  = f2bf(rscale * k2[t] * v2[t]);
            a[12 + t] = f2bf(rscale * k3[t] * v3[t]);
        }
#pragma unroll
        for (int c = 0; c < 4; ++c) {
            bf16x16 b = load_fragB(tgtT, d0 + c * 16, N_PTS, jb);
            acc[c] = wmma_bf16(a, b, acc[c]);
        }
    }
    int rbase = i0 + ((lane >> 4) << 3);
    float e2 = 0.f;
#pragma unroll
    for (int c = 0; c < 4; ++c) {
        int col = d0 + c * 16 + (lane & 15);
#pragma unroll
        for (int r = 0; r < 8; ++r) {
            float diff = src[(size_t)(rbase + r) * DIMS + col] - acc[c][r];
            e2 += diff * diff;
        }
    }
    e2 = wave_sum(e2);
    __shared__ float sred[8];
    if (lane == 0) sred[wib] = e2;
    __syncthreads();
    if (threadIdx.x == 0) {
        float s = 0.f;
        for (int w = 0; w < 8; ++w) s += sred[w];
        mse_part[blockIdx.x] = s;
    }
}

// --- ot partials: sum u[i]*K*v[j]*C ----------------------------------------
__global__ void k_ot(const float* __restrict__ Km, const float* __restrict__ C,
                     const float* __restrict__ u, const float* __restrict__ v,
                     float* __restrict__ ot_part) {
    size_t stride = (size_t)gridDim.x * blockDim.x;
    float acc = 0.f;
    for (size_t e = (size_t)blockIdx.x * blockDim.x + threadIdx.x;
         e < (size_t)N_PTS * N_PTS; e += stride) {
        int i = (int)(e >> 12);
        int j = (int)(e & 4095);
        acc += u[i] * Km[e] * v[j] * C[e];
    }
    __shared__ float sred[256];
    sred[threadIdx.x] = acc;
    __syncthreads();
    for (int s = 128; s > 0; s >>= 1) {
        if (threadIdx.x < s) sred[threadIdx.x] += sred[threadIdx.x + s];
        __syncthreads();
    }
    if (threadIdx.x == 0) ot_part[blockIdx.x] = sred[0];
}

// --- final: reduce partials, top-5 of row_min, combine ----------------------
__global__ void k_final(const float* __restrict__ ot_part, int n_ot,
                        const float* __restrict__ mse_part, int n_mse,
                        const float* __restrict__ row_min, float* __restrict__ out) {
    __shared__ float sred[256];
    __shared__ float bval[256];
    __shared__ int   bidx[256];
    __shared__ int   chosen[5];
    __shared__ float csum[3];
    int tid = threadIdx.x;

    float a = 0.f;
    for (int e = tid; e < n_ot; e += 256) a += ot_part[e];
    sred[tid] = a; __syncthreads();
    for (int s = 128; s > 0; s >>= 1) { if (tid < s) sred[tid] += sred[tid + s]; __syncthreads(); }
    if (tid == 0) csum[0] = sred[0];
    __syncthreads();

    a = 0.f;
    for (int e = tid; e < n_mse; e += 256) a += mse_part[e];
    sred[tid] = a; __syncthreads();
    for (int s = 128; s > 0; s >>= 1) { if (tid < s) sred[tid] += sred[tid + s]; __syncthreads(); }
    if (tid == 0) { csum[1] = sred[0]; csum[2] = 0.f; }
    __syncthreads();

    for (int t = 0; t < 5; ++t) {
        float bv = -3.4e38f; int bi = -1;
        for (int e = tid; e < N_PTS; e += 256) {
            bool skip = false;
            for (int c = 0; c < t; ++c) if (chosen[c] == e) skip = true;
            float vv = row_min[e];
            if (!skip && (vv > bv || (vv == bv && e < bi))) { bv = vv; bi = e; }
        }
        bval[tid] = bv; bidx[tid] = bi;
        __syncthreads();
        if (tid == 0) {
            float mb = bval[0]; int mi = bidx[0];
            for (int q = 1; q < 256; ++q)
                if (bval[q] > mb || (bval[q] == mb && bidx[q] >= 0 && bidx[q] < mi)) {
                    mb = bval[q]; mi = bidx[q];
                }
            chosen[t] = mi;
            csum[2] += mb;
        }
        __syncthreads();
    }
    if (tid == 0) {
        float ot   = csum[0];
        float mse  = csum[1] / (4096.0f * 256.0f);
        float topk = csum[2] / 5.0f;
        out[0] = 0.25f * ot + 0.5f * mse + 0.05f * topk;
    }
}

extern "C" void kernel_launch(void* const* d_in, const int* in_sizes, int n_in,
                              void* d_out, int out_size, void* d_ws, size_t ws_size,
                              hipStream_t stream) {
    (void)in_sizes; (void)n_in; (void)out_size; (void)ws_size;
    const float* src = (const float*)d_in[0];
    const float* tgt = (const float*)d_in[1];
    float* out = (float*)d_out;

    char* ws = (char*)d_ws;
    size_t off = 0;
    auto alloc = [&](size_t bytes) -> void* {
        void* p = ws + off;
        off = (off + bytes + 1023) & ~(size_t)1023;
        return p;
    };
    const size_t NN = (size_t)N_PTS * N_PTS;
    float*  C       = (float*)alloc(NN * 4);                 // 64 MB
    float*  Km      = (float*)alloc(NN * 4);                 // 64 MB
    __bf16* src_hi  = (__bf16*)alloc((size_t)N_PTS * DIMS * 2);
    __bf16* src_lo  = (__bf16*)alloc((size_t)N_PTS * DIMS * 2);
    __bf16* tgt_hi  = (__bf16*)alloc((size_t)N_PTS * DIMS * 2);
    __bf16* tgt_lo  = (__bf16*)alloc((size_t)N_PTS * DIMS * 2);
    __bf16* tgtT    = (__bf16*)alloc((size_t)DIMS * N_PTS * 2);
    float*  x2      = (float*)alloc(N_PTS * 4);
    float*  y2      = (float*)alloc(N_PTS * 4);
    float*  u       = (float*)alloc(N_PTS * 4);
    float*  v       = (float*)alloc(N_PTS * 4);
    float*  row_min = (float*)alloc(N_PTS * 4);
    float*  colpart = (float*)alloc((size_t)32 * N_PTS * 4); // 512 KB
    float*  ot_part = (float*)alloc(1024 * 4);
    float*  mse_part= (float*)alloc(128 * 4);

    k_pack  <<<4096, 256, 0, stream>>>(src, tgt, src_hi, src_lo, tgt_hi, tgt_lo, tgtT, u, v);
    k_norms <<<1024, 256, 0, stream>>>(src, tgt, x2, y2);
    k_gemm_C<<<2048, 256, 0, stream>>>(src_hi, src_lo, tgt_hi, tgt_lo, x2, y2, C, Km);
    k_rowmin<<<512, 256, 0, stream>>>(C, row_min);

    for (int it = 0; it < 50; ++it) {
        k_u_update<<<512, 256, 0, stream>>>(Km, v, u);
        k_colsum  <<<dim3(16, 32), 256, 0, stream>>>(Km, u, colpart);
        k_v_update<<<16, 256, 0, stream>>>(colpart, v);
        if (it == 19) {  // P_m state (20 iterations) — stream order snapshots u,v
            k_matched_mse<<<128, 256, 0, stream>>>(Km, u, v, tgtT, src, mse_part);
        }
    }
    k_ot   <<<1024, 256, 0, stream>>>(Km, C, u, v, ot_part);
    k_final<<<1, 256, 0, stream>>>(ot_part, 1024, mse_part, 128, row_min, out);
}
